// MSTAGNN_3083786518694
// MI455X (gfx1250) — matrix operand order repaired and checked by reading, not compiled, not run on hardware
//
#include <hip/hip_runtime.h>

// ---------------------------------------------------------------------------
// MSTAGNN gated temporal-attention GCN for MI455X (gfx1250, wave32, WMMA).
// All heavy matmuls run through v_wmma_f32_16x16x32_bf16 (fp32 accumulate).
// Requires ws_size >= ~104 MB.
// ---------------------------------------------------------------------------

constexpr int BB = 8, KK = 2, BT = 16;   // batch, time, flattened (b,t)
constexpr int NN = 2048;                 // nodes
constexpr int TT = 12;                   // history length
constexpr int DI = 32, DH = 64, DE = 24;
constexpr int CC = DI + DH;              // 96
constexpr int KI = 2 * CC;               // 192
constexpr int DEP = 32;                  // DE padded to 32 for K=32 WMMA
constexpr int JT = DH * KI;              // 12288 flattened (o,ki)

typedef __attribute__((ext_vector_type(8)))  __bf16 v8bf;
typedef __attribute__((ext_vector_type(16))) __bf16 v16bf;
typedef __attribute__((ext_vector_type(8)))  float  v8f;

__device__ inline v8f zerov8() {
  v8f z;
#pragma unroll
  for (int i = 0; i < 8; ++i) z[i] = 0.f;
  return z;
}

__device__ inline v8f wmma_bf16(v16bf a, v16bf b, v8f c) {
  // D = A(16x32 bf16) * B(32x16 bf16) + C(16x16 f32)
  return __builtin_amdgcn_wmma_f32_16x16x32_bf16(false, a, false, b,
                                                 (short)0, c, false, false);
}

// A-operand: 16 rows x 32 K-slice from row-major (row, ldK) bf16 memory.
// Layout: lanes 0-15 hold K{0..15} of row=lane; lanes 16-31 hold
// K{16..31} of row=lane-16.  One 32B load per lane.
__device__ inline v16bf load_a16x32(const __bf16* base, int ldK) {
  const int l = threadIdx.x & 31;
  const int row = l & 15;
  const int kh = (l < 16) ? 0 : 16;
  const __bf16* p = base + (size_t)row * ldK + kh;
  return *(const v16bf*)(p);
}

// B-operand: 32(K) x 16(N) tile; memory holds it transposed (N rows, K inner).
// Lanes 0-15: N=lane, K=0..15; lanes 16-31: N=lane-16, K=16..31. One 32B load.
__device__ inline v16bf load_b32x16(const __bf16* base, int ldK) {
  const int l = threadIdx.x & 31;
  const int col = l & 15;
  const int k0 = (l < 16) ? 0 : 16;
  return *(const v16bf*)(base + (size_t)col * ldK + k0);
}

// ---------------------------------------------------------------------------
// Prep kernels
// ---------------------------------------------------------------------------

// emb[g][bt][n][0..31] = LN_{24}(node_emb[n]+time_emb[bt]) * lnA_g + lnA_b (bf16, zero-pad)
__global__ void __launch_bounds__(256)
prep_emb_kernel(const float* __restrict__ ne, const float* __restrict__ te,
                const float* __restrict__ g0, const float* __restrict__ b0,
                const float* __restrict__ g1, const float* __restrict__ b1,
                const float* __restrict__ g2, const float* __restrict__ b2,
                __bf16* __restrict__ emb) {
  int tid = blockIdx.x * 256 + threadIdx.x;
  if (tid >= BT * NN) return;
  int n = tid % NN, bt = tid / NN;
  float s[DE];
  float mean = 0.f;
#pragma unroll
  for (int d = 0; d < DE; ++d) { s[d] = ne[n * DE + d] + te[bt * DE + d]; mean += s[d]; }
  mean *= (1.f / DE);
  float var = 0.f;
#pragma unroll
  for (int d = 0; d < DE; ++d) { float t = s[d] - mean; var += t * t; }
  var *= (1.f / DE);
  float rinv = rsqrtf(var + 1e-12f);
  const float* gs[3] = {g0, g1, g2};
  const float* bs[3] = {b0, b1, b2};
  for (int gi = 0; gi < 3; ++gi) {
    __bf16* dst = emb + ((size_t)(gi * BT + bt) * NN + n) * DEP;
#pragma unroll
    for (int d = 0; d < DE; ++d)
      dst[d] = (__bf16)((s[d] - mean) * rinv * gs[gi][d] + bs[gi][d]);
#pragma unroll
    for (int d = DE; d < DEP; ++d) dst[d] = (__bf16)0.f;
  }
}

// node_emb -> bf16, zero-padded to 32 (A-operand of the W GEMM)
__global__ void __launch_bounds__(256)
prep_ne_kernel(const float* __restrict__ ne, __bf16* __restrict__ nebf) {
  int tid = blockIdx.x * 256 + threadIdx.x;
  if (tid >= NN * DEP) return;
  int d = tid % DEP, n = tid / DEP;
  nebf[tid] = (d < DE) ? (__bf16)ne[n * DE + d] : (__bf16)0.f;
}

// WpT[g][j = o*192 + k*96 + i][d(32)] = Wp_g[d][k][i][o]  (B-operand of W GEMM)
__global__ void __launch_bounds__(256)
prep_wpt_kernel(const float* __restrict__ Wz, const float* __restrict__ Wr,
                const float* __restrict__ Wu, __bf16* __restrict__ wpt) {
  int tid = blockIdx.x * 256 + threadIdx.x;
  if (tid >= 3 * JT * DEP) return;
  int d = tid % DEP;
  int j = (tid / DEP) % JT;
  int g = tid / (JT * DEP);
  int o = j / KI, ki = j % KI, k = ki / CC, i = ki % CC;
  const float* W = (g == 0) ? Wz : ((g == 1) ? Wr : Wu);
  float v = (d < DE) ? W[(((size_t)d * 2 + k) * CC + i) * DH + o] : 0.f;
  wpt[tid] = (__bf16)v;
}

// bias[g][bt][o] = time_emb[bt] . bp_g[:,o]
__global__ void __launch_bounds__(256)
prep_bias_kernel(const float* __restrict__ te, const float* __restrict__ bz,
                 const float* __restrict__ br, const float* __restrict__ bu,
                 float* __restrict__ bias) {
  int tid = blockIdx.x * 256 + threadIdx.x;
  if (tid >= 3 * BT * DH) return;
  int o = tid % DH;
  int bt = (tid / DH) % BT;
  int g = tid / (BT * DH);
  const float* bp = (g == 0) ? bz : ((g == 1) ? br : bu);
  float acc = 0.f;
#pragma unroll
  for (int d = 0; d < DE; ++d) acc += te[bt * DE + d] * bp[d * DH + o];
  bias[tid] = acc;
}

// xg[n][bt][c] (c<96) and xsT[bt][c][n] = concat(x, zf*state) in bf16.
__global__ void __launch_bounds__(256)
pack_kernel(const float* __restrict__ x, const float* __restrict__ states,
            const float* __restrict__ zf, __bf16* __restrict__ xg,
            __bf16* __restrict__ xsT) {
  int tid = blockIdx.x * 256 + threadIdx.x;
  if (tid >= BT * NN * CC) return;
  int c = tid % CC;
  int n = (tid / CC) % NN;
  int bt = tid / (CC * NN);
  int b = bt >> 1, t = bt & 1;
  float v;
  if (c < DI) {
    v = x[((size_t)bt * NN + n) * DI + c];
  } else {
    float sv = states[(((size_t)b * TT + (TT - KK) + t) * NN + n) * DH + (c - DI)];
    float f = zf ? zf[((size_t)bt * NN + n) * DH + (c - DI)] : 1.0f;
    v = f * sv;
  }
  __bf16 bv = (__bf16)v;
  xg[((size_t)n * BT + bt) * KI + c] = bv;
  xsT[((size_t)bt * CC + c) * NN + n] = bv;
}

// ---------------------------------------------------------------------------
// Flash node-attention: softmax(emb_rows . emb_cols^T) @ xs  -> xg[:, :, 96:192]
// grid (N/128, BT), 256 threads (8 waves x 16 rows each).
// Ping-pong LDS staging: one barrier per 64-column chunk; next chunk's global
// load is issued before compute so its latency hides under 16 WMMAs+softmax.
// ---------------------------------------------------------------------------
__global__ void __launch_bounds__(256)
flash_kernel(const __bf16* __restrict__ embG,  // [BT][N][32]
             const __bf16* __restrict__ xsT,   // [BT][96][N]
             __bf16* __restrict__ xg) {        // [N][BT][192]
  __shared__ __align__(32) __bf16 sm_cols[2][64 * DEP];  // 2 x 4 KB ping-pong
  __shared__ __align__(32) __bf16 sm_P[8][16 * 64];      // 16 KB
  const int bt = blockIdx.y;
  const int wave = threadIdx.x >> 5;
  const int lane = threadIdx.x & 31;
  const int row0 = blockIdx.x * 128 + wave * 16;

  const __bf16* emb_bt = embG + (size_t)bt * NN * DEP;
  const __bf16* xsT_bt = xsT + (size_t)bt * CC * NN;

  const v16bf a_emb = load_a16x32(emb_bt + (size_t)row0 * DEP, DEP);

  float mrow[8], lrow[8];
  v8f O[6];
#pragma unroll
  for (int r = 0; r < 8; ++r) { mrow[r] = -3.0e38f; lrow[r] = 0.f; }
#pragma unroll
  for (int c = 0; c < 6; ++c) O[c] = zerov8();

  // prologue: stage chunk 0 into buffer 0
  *(v8bf*)(&sm_cols[0][threadIdx.x * 8]) =
      *(const v8bf*)(emb_bt + threadIdx.x * 8);
  __syncthreads();

  for (int mc = 0; mc < NN / 64; ++mc) {
    const int p = mc & 1;
    const bool have_next = (mc + 1) < (NN / 64);

    // issue next chunk's staging load early (latency hidden under compute)
    v8bf nxt;
    if (have_next) {
      nxt = *(const v8bf*)(emb_bt + (size_t)(mc + 1) * 64 * DEP +
                           threadIdx.x * 8);
      // pull next PV K-panel (96 rows x 128B) toward the WGP
      if (threadIdx.x < CC)
        __builtin_prefetch(xsT_bt + (size_t)threadIdx.x * NN + (mc + 1) * 64,
                           0, 1);
    }

    // S = emb_rows @ emb_cols^T : 4 tiles of 16 columns, K=32 (DE padded)
    v8f S[4];
#pragma unroll
    for (int mt = 0; mt < 4; ++mt) {
      v16bf b = load_b32x16(&sm_cols[p][mt * 16 * DEP], DEP);
      S[mt] = wmma_bf16(a_emb, b, zerov8());
    }

    // online softmax (row r lives in a 16-lane half; width-16 xor reductions)
    float nm[8], alpha[8], rs[8];
#pragma unroll
    for (int r = 0; r < 8; ++r) {
      float cm = S[0][r];
#pragma unroll
      for (int mt = 1; mt < 4; ++mt) cm = fmaxf(cm, S[mt][r]);
#pragma unroll
      for (int sh = 8; sh >= 1; sh >>= 1) cm = fmaxf(cm, __shfl_xor(cm, sh, 16));
      nm[r] = fmaxf(mrow[r], cm);
      alpha[r] = __expf(mrow[r] - nm[r]);
      mrow[r] = nm[r];
      rs[r] = 0.f;
    }

    // P = exp(S - m), stash row-major 16x64 in wave-private LDS (A relayout)
    {
      const int colb = lane & 15;
      const int rofs = (lane < 16) ? 0 : 8;
#pragma unroll
      for (int mt = 0; mt < 4; ++mt) {
#pragma unroll
        for (int r = 0; r < 8; ++r) {
          float pv = __expf(S[mt][r] - nm[r]);
          rs[r] += pv;
          sm_P[wave][(r + rofs) * 64 + mt * 16 + colb] = (__bf16)pv;
        }
      }
    }
#pragma unroll
    for (int r = 0; r < 8; ++r) {
#pragma unroll
      for (int sh = 8; sh >= 1; sh >>= 1) rs[r] += __shfl_xor(rs[r], sh, 16);
      lrow[r] = lrow[r] * alpha[r] + rs[r];
    }
#pragma unroll
    for (int c = 0; c < 6; ++c)
#pragma unroll
      for (int r = 0; r < 8; ++r) O[c][r] *= alpha[r];

    asm volatile("s_wait_dscnt 0" ::: "memory");  // P stores visible to reads

    // O += P(16x64) @ xs_chunk(64x96): 2 K-steps x 6 channel tiles
#pragma unroll
    for (int ks = 0; ks < 2; ++ks) {
      v16bf aP = load_a16x32(&sm_P[wave][ks * 32], 64);
      const int m0 = mc * 64 + ks * 32;
#pragma unroll
      for (int ct = 0; ct < 6; ++ct) {
        v16bf b = load_b32x16(xsT_bt + (size_t)(ct * 16) * NN + m0, NN);
        O[ct] = wmma_bf16(aP, b, O[ct]);
      }
    }

    // publish next chunk into the other buffer; single barrier per iteration
    if (have_next)
      *(v8bf*)(&sm_cols[p ^ 1][threadIdx.x * 8]) = nxt;
    __syncthreads();
  }

  // normalize and scatter into xg[n][bt][96 + c]
#pragma unroll
  for (int r = 0; r < 8; ++r) {
    float inv = 1.0f / lrow[r];
    int n = row0 + r + ((lane < 16) ? 0 : 8);
    __bf16* dst = xg + ((size_t)n * BT + bt) * KI + CC;
#pragma unroll
    for (int ct = 0; ct < 6; ++ct)
      dst[ct * 16 + (lane & 15)] = (__bf16)(O[ct][r] * inv);
  }
}

// ---------------------------------------------------------------------------
// W GEMM: Wt[n][o][ki] = node_emb[n,:] . Wp[:, ki, o]   (M=2048, N=12288, K=32)
// grid (JT/64, NN/128), 256 threads.
// ---------------------------------------------------------------------------
__global__ void __launch_bounds__(256)
gemmw_kernel(const __bf16* __restrict__ nebf,  // [N][32]
             const __bf16* __restrict__ wpt,   // [12288][32]
             __bf16* __restrict__ wt) {        // [N][12288]
  const int wave = threadIdx.x >> 5;
  const int lane = threadIdx.x & 31;
  const int row0 = blockIdx.y * 128 + wave * 16;
  const int j0 = blockIdx.x * 64;
  const v16bf a = load_a16x32(nebf + (size_t)row0 * DEP, DEP);
#pragma unroll
  for (int jt = 0; jt < 4; ++jt) {
    v16bf b = load_b32x16(wpt + (size_t)(j0 + jt * 16) * DEP, DEP);
    v8f c = wmma_bf16(a, b, zerov8());
    const int col = j0 + jt * 16 + (lane & 15);
    const int rofs = (lane < 16) ? 0 : 8;
#pragma unroll
    for (int r = 0; r < 8; ++r)
      wt[(size_t)(row0 + r + rofs) * JT + col] = (__bf16)c[r];
  }
}

// ---------------------------------------------------------------------------
// Per-node apply: g[bt][n][o] = xg[n](16x192) @ Wt[n](192x64) + bias[bt][o]
// One wave per node; 256 blocks x 8 waves.
// ---------------------------------------------------------------------------
__global__ void __launch_bounds__(256)
apply_kernel(const __bf16* __restrict__ xg,   // [N][16][192]
             const __bf16* __restrict__ wt,   // [N][64][192]
             const float* __restrict__ bias,  // [16][64]
             float* __restrict__ g) {         // [16][N][64]
  const int wave = threadIdx.x >> 5;
  const int lane = threadIdx.x & 31;
  const int n = blockIdx.x * 8 + wave;
  const __bf16* A = xg + (size_t)n * BT * KI;
  const __bf16* Bm = wt + (size_t)n * DH * KI;
  v8f acc[4];
#pragma unroll
  for (int ot = 0; ot < 4; ++ot) acc[ot] = zerov8();
#pragma unroll
  for (int ks = 0; ks < 6; ++ks) {
    v16bf a = load_a16x32(A + ks * 32, KI);
#pragma unroll
    for (int ot = 0; ot < 4; ++ot) {
      v16bf b = load_b32x16(Bm + (size_t)(ot * 16) * KI + ks * 32, KI);
      acc[ot] = wmma_bf16(a, b, acc[ot]);
    }
  }
  const int rofs = (lane < 16) ? 0 : 8;
#pragma unroll
  for (int ot = 0; ot < 4; ++ot) {
    const int o = ot * 16 + (lane & 15);
#pragma unroll
    for (int r = 0; r < 8; ++r) {
      const int bt = r + rofs;
      g[((size_t)bt * NN + n) * DH + o] = acc[ot][r] + bias[bt * DH + o];
    }
  }
}

// ---------------------------------------------------------------------------
// MHA over history + LN + activation, in place on g.
// thread per (b,t,n): q = LN(g row), 4 heads x 16, s = 12 keys.
// ---------------------------------------------------------------------------
__global__ void __launch_bounds__(128)
mha_kernel(float* __restrict__ g, const float* __restrict__ states,
           const float* __restrict__ lng, const float* __restrict__ lnb,
           int actmode) {
  int tid = blockIdx.x * 128 + threadIdx.x;
  if (tid >= BT * NN) return;
  int n = tid % NN, bt = tid / NN;
  int b = bt >> 1;
  float* grow = g + ((size_t)bt * NN + n) * DH;
  float gv[DH];
  float mean = 0.f;
#pragma unroll
  for (int o = 0; o < DH; ++o) { gv[o] = grow[o]; mean += gv[o]; }
  mean *= (1.f / DH);
  float var = 0.f;
#pragma unroll
  for (int o = 0; o < DH; ++o) { float t = gv[o] - mean; var += t * t; }
  var *= (1.f / DH);
  float rinv = rsqrtf(var + 1e-5f);
  float qn[DH];
#pragma unroll
  for (int o = 0; o < DH; ++o) qn[o] = (gv[o] - mean) * rinv * lng[o] + lnb[o];

  const size_t kvstride = (size_t)NN * DH;  // one history step
  const float* kv0 = states + ((size_t)b * TT * NN + n) * DH;
  for (int h = 0; h < 4; ++h) {
    float sc[TT];
    float mx = -3.0e38f;
#pragma unroll
    for (int s = 0; s < TT; ++s) {
      const float* kp = kv0 + (size_t)s * kvstride + h * 16;
      float dot = 0.f;
#pragma unroll
      for (int j = 0; j < 16; ++j) dot += qn[h * 16 + j] * kp[j];
      sc[s] = dot * 0.25f;  // 1/sqrt(16)
      mx = fmaxf(mx, sc[s]);
    }
    float se = 0.f;
#pragma unroll
    for (int s = 0; s < TT; ++s) { sc[s] = __expf(sc[s] - mx); se += sc[s]; }
    float sinv = 1.f / se;
#pragma unroll
    for (int j = 0; j < 16; ++j) {
      float acc = 0.f;
#pragma unroll
      for (int s = 0; s < TT; ++s)
        acc += sc[s] * kv0[(size_t)s * kvstride + h * 16 + j];
      float val = gv[h * 16 + j] + acc * sinv;
      grow[h * 16 + j] =
          (actmode == 0) ? 1.f / (1.f + __expf(-val)) : tanhf(val);
    }
  }
}

// out = r*state + (1-r)*hc
__global__ void __launch_bounds__(256)
final_kernel(const float* __restrict__ r, const float* __restrict__ hc,
             const float* __restrict__ states, float* __restrict__ out) {
  int tid = blockIdx.x * 256 + threadIdx.x;
  if (tid >= BT * NN * DH) return;
  int o = tid % DH;
  int n = (tid / DH) % NN;
  int bt = tid / (DH * NN);
  int b = bt >> 1, t = bt & 1;
  float st = states[(((size_t)b * TT + (TT - KK) + t) * NN + n) * DH + o];
  float rv = r[tid];
  out[tid] = rv * st + (1.f - rv) * hc[tid];
}

// ---------------------------------------------------------------------------
// Host orchestration
// ---------------------------------------------------------------------------
constexpr size_t OFF_EMB  = 0;                                     // 6,291,456
constexpr size_t OFF_NE   = OFF_EMB + 3ull * BT * NN * DEP * 2;    //   131,072
constexpr size_t OFF_WPT  = OFF_NE + (size_t)NN * DEP * 2;         // 2,359,296
constexpr size_t OFF_XST  = OFF_WPT + 3ull * JT * DEP * 2;         // 6,291,456
constexpr size_t OFF_XG   = OFF_XST + (size_t)BT * CC * NN * 2;    // 12,582,912
constexpr size_t OFF_WT   = OFF_XG + (size_t)NN * BT * KI * 2;     // 50,331,648
constexpr size_t OFF_G    = OFF_WT + (size_t)NN * DH * KI * 2;     // 25,165,824
constexpr size_t OFF_BIAS = OFF_G + 3ull * BT * NN * DH * 4;       //    12,288

extern "C" void kernel_launch(void* const* d_in, const int* in_sizes, int n_in,
                              void* d_out, int out_size, void* d_ws,
                              size_t ws_size, hipStream_t stream) {
  const float* x      = (const float*)d_in[0];
  const float* states = (const float*)d_in[1];
  const float* ne     = (const float*)d_in[2];
  const float* te     = (const float*)d_in[3];
  const float* Wg[3]  = {(const float*)d_in[4], (const float*)d_in[10], (const float*)d_in[16]};
  const float* bp[3]  = {(const float*)d_in[5], (const float*)d_in[11], (const float*)d_in[17]};
  const float* lnAg[3] = {(const float*)d_in[6], (const float*)d_in[12], (const float*)d_in[18]};
  const float* lnAb[3] = {(const float*)d_in[7], (const float*)d_in[13], (const float*)d_in[19]};
  const float* lnOg[3] = {(const float*)d_in[8], (const float*)d_in[14], (const float*)d_in[20]};
  const float* lnOb[3] = {(const float*)d_in[9], (const float*)d_in[15], (const float*)d_in[21]};

  char* ws = (char*)d_ws;
  __bf16* EMB = (__bf16*)(ws + OFF_EMB);
  __bf16* NE  = (__bf16*)(ws + OFF_NE);
  __bf16* WPT = (__bf16*)(ws + OFF_WPT);
  __bf16* XST = (__bf16*)(ws + OFF_XST);
  __bf16* XG  = (__bf16*)(ws + OFF_XG);
  __bf16* WT  = (__bf16*)(ws + OFF_WT);
  float*  BIAS = (float*)(ws + OFF_BIAS);
  float* G[3];
  for (int i = 0; i < 3; ++i) G[i] = (float*)(ws + OFF_G) + (size_t)i * BT * NN * DH;

  // ---- prep (independent) ----
  prep_emb_kernel<<<128, 256, 0, stream>>>(ne, te, lnAg[0], lnAb[0], lnAg[1],
                                           lnAb[1], lnAg[2], lnAb[2], EMB);
  prep_ne_kernel<<<256, 256, 0, stream>>>(ne, NE);
  prep_wpt_kernel<<<4608, 256, 0, stream>>>(Wg[0], Wg[1], Wg[2], WPT);
  prep_bias_kernel<<<12, 256, 0, stream>>>(te, bp[0], bp[1], bp[2], BIAS);

  // ---- phase A: z (gate 0) and r (gate 1) on xs = [x, state] ----
  pack_kernel<<<12288, 256, 0, stream>>>(x, states, nullptr, XG, XST);
  for (int gi = 0; gi < 2; ++gi) {
    flash_kernel<<<dim3(NN / 128, BT), 256, 0, stream>>>(
        EMB + (size_t)gi * BT * NN * DEP, XST, XG);
    gemmw_kernel<<<dim3(JT / 64, NN / 128), 256, 0, stream>>>(
        NE, WPT + (size_t)gi * JT * DEP, WT);
    apply_kernel<<<NN / 8, 256, 0, stream>>>(XG, WT, BIAS + gi * BT * DH, G[gi]);
    mha_kernel<<<256, 128, 0, stream>>>(G[gi], states, lnOg[gi], lnOb[gi], 0);
  }

  // ---- phase B: u (gate 2) on cand = [x, z*state] ----
  pack_kernel<<<12288, 256, 0, stream>>>(x, states, G[0], XG, XST);
  flash_kernel<<<dim3(NN / 128, BT), 256, 0, stream>>>(
      EMB + 2ull * BT * NN * DEP, XST, XG);
  gemmw_kernel<<<dim3(JT / 64, NN / 128), 256, 0, stream>>>(
      NE, WPT + 2ull * JT * DEP, WT);
  apply_kernel<<<NN / 8, 256, 0, stream>>>(XG, WT, BIAS + 2 * BT * DH, G[2]);
  mha_kernel<<<256, 128, 0, stream>>>(G[2], states, lnOg[2], lnOb[2], 1);

  // ---- combine ----
  final_kernel<<<8192, 256, 0, stream>>>(G[1], G[2], states, (float*)d_out);
}
